// VisualHead_79714593014222
// MI455X (gfx1250) — matrix operand order repaired and checked by reading, not compiled, not run on hardware
//
#include <hip/hip_runtime.h>
#include <stdint.h>

#define B_   16
#define T_   512
#define D_   1024
#define FF_  2048
#define E_   4
#define CLS_ 1200
#define KW_  3
#define NTOK (B_*T_)          // 8192

// ---- trivial (union-safe) vector types ----
typedef __attribute__((ext_vector_type(16))) __bf16 v16bf;
typedef __attribute__((ext_vector_type(8)))  float  v8f;
typedef __attribute__((ext_vector_type(4)))  unsigned int u32x4;
typedef __attribute__((ext_vector_type(4)))  float  f32x4;

union FragAB { u32x4 u[2]; v16bf v; };

__device__ __forceinline__ unsigned short f2bs(float f) {
  union { float f; unsigned int u; } a; a.f = f;
  unsigned int r = a.u + 0x7FFFu + ((a.u >> 16) & 1u);   // RNE
  return (unsigned short)(r >> 16);
}
__device__ __forceinline__ float bs2f(unsigned short s) {
  union { unsigned int u; float f; } a; a.u = ((unsigned int)s) << 16;
  return a.f;
}

__device__ __forceinline__ v8f wmma_bf16(v16bf a, v16bf b, v8f c) {
  return __builtin_amdgcn_wmma_f32_16x16x32_bf16(false, a, false, b, (short)0, c, false, false);
}

// ---- CDNA5 async global->LDS DMA (ASYNCcnt-tracked), 16 B per active lane ----
__device__ __forceinline__ void async_ld_b128(unsigned lds_byte_addr, const void* gaddr) {
  asm volatile("global_load_async_to_lds_b128 %0, %1, off"
               :: "v"(lds_byte_addr), "v"(gaddr) : "memory");
}
__device__ __forceinline__ void async_wait0() {
  asm volatile("s_wait_asynccnt 0" ::: "memory");
}

// ---- LDS tile geometry (row stride 40 shorts = 80B = 16B aligned, bank-staggered) ----
#define LDSA_S 40
#define LDSB_S 40

__device__ __forceinline__ v16bf load_fragA(const unsigned short* lA, int wm, int mr, int lane) {
  int g = lane >> 4, lr = lane & 15;
  int row = wm*32 + mr*16 + lr;
  FragAB f;
  f.u[0] = *(const u32x4*)&lA[row*LDSA_S + g*8];        // K = g*8 .. g*8+7
  f.u[1] = *(const u32x4*)&lA[row*LDSA_S + 16 + g*8];   // K = 16+g*8 ..
  return f.v;
}
__device__ __forceinline__ v16bf load_fragB(const unsigned short* lB, int wn, int nf, int lane) {
  int g = lane >> 4, lr = lane & 15;
  int n = wn*32 + nf*16 + lr;
  FragAB f;
  f.u[0] = *(const u32x4*)&lB[n*LDSB_S + g*8];          // K = g*8 .. g*8+7
  f.u[1] = *(const u32x4*)&lB[n*LDSB_S + 16 + g*8];     // K = 16+g*8 ..
  return f.v;
}

// =======================================================================
// Weight packing (fp32 -> bf16, conv taps split out as plain GEMM A mats)
// =======================================================================
__global__ void k_cvt_bf16(const float* __restrict__ src, unsigned short* __restrict__ dst, int n) {
  int i = blockIdx.x*blockDim.x + threadIdx.x;
  if (i < n) dst[i] = f2bs(src[i]);
}
__global__ void k_pack_w1(const float* __restrict__ ew1, unsigned short* __restrict__ w1p) {
  // dst[((e*KW+k)*FF+f)*D+d] = ew1[((e*FF+f)*D+d)*KW+k]
  int i = blockIdx.x*blockDim.x + threadIdx.x;
  int d = i % D_;  int t = i / D_;
  int f = t % FF_; t /= FF_;
  int k = t % KW_; int e = t / KW_;
  w1p[i] = f2bs(ew1[(((size_t)e*FF_ + f)*D_ + d)*KW_ + k]);
}
__global__ void k_pack_w2(const float* __restrict__ ew2, unsigned short* __restrict__ w2p) {
  // dst[((e*KW+k)*D+d)*FF+f] = ew2[((e*D+d)*FF+f)*KW+k]
  int i = blockIdx.x*blockDim.x + threadIdx.x;
  int f = i % FF_; int t = i / FF_;
  int d = t % D_;  t /= D_;
  int k = t % KW_; int e = t / KW_;
  w2p[i] = f2bs(ew2[(((size_t)e*D_ + d)*FF_ + f)*KW_ + k]);
}

// =======================================================================
// fc1: H[token][m] = sum_d X[token][d] * W[m][d] + b[m]
// A (weights) via async DMA; B converts f32->bf16 through registers.
// Double-buffered LDS, one barrier per K-step.
// =======================================================================
__global__ __launch_bounds__(256) void k_gemm_fc1(
    const unsigned short* __restrict__ Wb,  // [D][D] bf16
    const float* __restrict__ X,            // [NTOK][D] f32
    const float* __restrict__ bias,         // [D]
    float* __restrict__ H)                  // [NTOK][D]
{
  __shared__ unsigned short lA[2][128*LDSA_S];
  __shared__ unsigned short lB[2][64*LDSB_S];
  int tid = threadIdx.x, lane = tid & 31, wave = tid >> 5;
  int wm = wave & 3, wn = wave >> 2;
  int n0 = blockIdx.x * 64;    // token tile
  int m0 = blockIdx.y * 128;   // channel tile
  unsigned baseA = (unsigned)(uintptr_t)&lA[0][0];
  const int arow = tid >> 1, ach = (tid & 1)*16;
  const int bn = tid >> 2,  bseg = (tid & 3)*8;
  const unsigned short* Arow = &Wb[(size_t)(m0 + arow)*D_];
  const float*          Brow = &X [(size_t)(n0 + bn)*D_];

  f32x4 br0, br1;
  auto issueA = [&](int k0, int buf) {
    unsigned la = baseA + (unsigned)((buf*128 + arow)*LDSA_S + ach)*2u;
    async_ld_b128(la,      Arow + k0 + ach);
    async_ld_b128(la + 16, Arow + k0 + ach + 8);
  };
  auto loadBreg = [&](int k0) {
    const f32x4* s = (const f32x4*)(Brow + k0 + bseg);
    br0 = s[0]; br1 = s[1];
  };
  auto storeB = [&](int buf) {
    u32x4 u;
    u[0] = (unsigned)f2bs(br0[0]) | ((unsigned)f2bs(br0[1]) << 16);
    u[1] = (unsigned)f2bs(br0[2]) | ((unsigned)f2bs(br0[3]) << 16);
    u[2] = (unsigned)f2bs(br1[0]) | ((unsigned)f2bs(br1[1]) << 16);
    u[3] = (unsigned)f2bs(br1[2]) | ((unsigned)f2bs(br1[3]) << 16);
    *(u32x4*)&lB[buf][bn*LDSB_S + bseg] = u;
  };

  v8f acc[2][2] = {};
  issueA(0, 0); loadBreg(0); storeB(0);
  async_wait0();
  __syncthreads();
  const int NS = D_/32;
  for (int st = 0; st < NS; ++st) {
    int cur = st & 1, nxt = cur ^ 1;
    if (st + 1 < NS) { issueA((st+1)*32, nxt); loadBreg((st+1)*32); }
    const unsigned short* cA = lA[cur];
    const unsigned short* cB = lB[cur];
    v16bf a0 = load_fragA(cA, wm, 0, lane), a1 = load_fragA(cA, wm, 1, lane);
    v16bf b0 = load_fragB(cB, wn, 0, lane), b1 = load_fragB(cB, wn, 1, lane);
    acc[0][0] = wmma_bf16(a0, b0, acc[0][0]);
    acc[0][1] = wmma_bf16(a0, b1, acc[0][1]);
    acc[1][0] = wmma_bf16(a1, b0, acc[1][0]);
    acc[1][1] = wmma_bf16(a1, b1, acc[1][1]);
    if (st + 1 < NS) { storeB(nxt); async_wait0(); }
    __syncthreads();
  }
  int g = lane >> 4, lr = lane & 15;
  for (int mr = 0; mr < 2; ++mr)
    for (int nf = 0; nf < 2; ++nf) {
      int m = m0 + wm*32 + mr*16 + g*8;
      int n = n0 + wn*32 + nf*16 + lr;
      f32x4 c0 = *(const f32x4*)&bias[m];
      f32x4 c1 = *(const f32x4*)&bias[m+4];
      v8f v = acc[mr][nf];
      f32x4 o0 = { v[0]+c0[0], v[1]+c0[1], v[2]+c0[2], v[3]+c0[3] };
      f32x4 o1 = { v[4]+c1[0], v[5]+c1[1], v[6]+c1[2], v[7]+c1[3] };
      float* o = &H[(size_t)n*D_ + m];
      *(f32x4*)&o[0] = o0;
      *(f32x4*)&o[4] = o1;
    }
}

// =======================================================================
// Masked batch-norm stats + apply + ReLU + PE  -> bf16 activations
// =======================================================================
__global__ void k_maskcnt(const int* __restrict__ mask, float* __restrict__ cnt) {
  __shared__ float r[256];
  int tid = threadIdx.x; float s = 0.f;
  for (int i = tid; i < NTOK; i += 256) s += (float)mask[i];
  r[tid] = s; __syncthreads();
  for (int st = 128; st > 0; st >>= 1) { if (tid < st) r[tid] += r[tid+st]; __syncthreads(); }
  if (tid == 0) cnt[0] = fmaxf(r[0], 1.0f);
}

__global__ __launch_bounds__(256) void k_stats(
    const float* __restrict__ H, const int* __restrict__ mask, float* __restrict__ stats /*[2][D]*/)
{
  int c  = blockIdx.x*256 + threadIdx.x;
  int t0 = blockIdx.y * 256;
  float s = 0.f, sq = 0.f;
  for (int t = t0; t < t0 + 256; ++t) {
    if (mask[t]) { float v = H[(size_t)t*D_ + c]; s += v; sq += v*v; }
  }
  atomicAdd(&stats[c], s);
  atomicAdd(&stats[D_ + c], sq);
}

__global__ void k_normapply(
    const float* __restrict__ H, const int* __restrict__ mask,
    const float* __restrict__ stats, const float* __restrict__ cnt,
    const float* __restrict__ bn_g, const float* __restrict__ bn_b,
    const float* __restrict__ pe, unsigned short* __restrict__ Hact)
{
  int i  = blockIdx.x*blockDim.x + threadIdx.x;  // NTOK*D
  int d  = i & (D_-1);
  int bt = i >> 10;
  int t  = bt & (T_-1);
  float c    = cnt[0];
  float mean = stats[d] / c;
  float var  = stats[D_ + d] / c - mean*mean;
  float h    = H[i];
  float nm   = (h - mean) * rsqrtf(var + 1e-5f) * bn_g[d] + bn_b[d];
  float v    = mask[bt] ? nm : h;
  v = fmaxf(v, 0.0f) + pe[(size_t)t*D_ + d];
  Hact[i] = f2bs(v);
}

// =======================================================================
// Gate: softmax + top-2 + aux-loss accumulators
// =======================================================================
__global__ __launch_bounds__(256) void k_gate(
    const unsigned short* __restrict__ Hact, const float* __restrict__ gw /*[D][E]*/,
    float* __restrict__ gates /*[NTOK][E]*/, float* __restrict__ aux /*[0..3]=imp [4..7]=cnt*/)
{
  int bt = blockIdx.x, tid = threadIdx.x;
  float p0=0,p1=0,p2=0,p3=0;
  for (int d = tid; d < D_; d += 256) {
    float h = bs2f(Hact[(size_t)bt*D_ + d]);
    f32x4 w = *(const f32x4*)&gw[d*4];
    p0 += h*w[0]; p1 += h*w[1]; p2 += h*w[2]; p3 += h*w[3];
  }
  __shared__ float red[4][256];
  red[0][tid]=p0; red[1][tid]=p1; red[2][tid]=p2; red[3][tid]=p3;
  __syncthreads();
  for (int s = 128; s > 0; s >>= 1) {
    if (tid < s) { red[0][tid]+=red[0][tid+s]; red[1][tid]+=red[1][tid+s];
                   red[2][tid]+=red[2][tid+s]; red[3][tid]+=red[3][tid+s]; }
    __syncthreads();
  }
  if (tid == 0) {
    float l[4] = { red[0][0], red[1][0], red[2][0], red[3][0] };
    float mx = fmaxf(fmaxf(l[0],l[1]), fmaxf(l[2],l[3]));
    float ex[4], se = 0.f;
    for (int e = 0; e < 4; ++e) { ex[e] = __expf(l[e]-mx); se += ex[e]; }
    int i1 = 0; for (int e = 1; e < 4; ++e) if (l[e] > l[i1]) i1 = e;
    int i2 = -1; for (int e = 0; e < 4; ++e) { if (e == i1) continue; if (i2 < 0 || l[e] > l[i2]) i2 = e; }
    float eb = __expf(l[i2]-l[i1]);
    float g1 = 1.0f/(1.0f+eb), g2 = eb/(1.0f+eb);
    float gv[4] = {0,0,0,0}; gv[i1] = g1; gv[i2] = g2;
    for (int e = 0; e < 4; ++e) {
      gates[(size_t)bt*4 + e] = gv[e];
      atomicAdd(&aux[e], ex[e]/se);
      if (gv[e] > 0.0f) atomicAdd(&aux[4+e], 1.0f);
    }
  }
}

// =======================================================================
// Expert conv1 (D->FF, KW=3): 3 accumulated WMMA GEMMs, async DMA tiles,
// double-buffered LDS; ReLU+bias; bf16 output.
// =======================================================================
__global__ __launch_bounds__(256) void k_conv1(
    const unsigned short* __restrict__ W1p,  // [E][KW][FF][D]
    const unsigned short* __restrict__ Hact, // [B][T][D]
    const float* __restrict__ eb1,           // [E][FF]
    unsigned short* __restrict__ Y1)         // [E][B][T][FF]
{
  __shared__ unsigned short lA[2][128*LDSA_S];
  __shared__ unsigned short lB[2][64*LDSB_S];
  int tid = threadIdx.x, lane = tid & 31, wave = tid >> 5;
  int wm = wave & 3, wn = wave >> 2;
  int nt = blockIdx.x & 7, mt = blockIdx.x >> 3;
  int b = blockIdx.y, e = blockIdx.z;
  int n0 = nt*64, m0 = mt*128;
  unsigned baseA = (unsigned)(uintptr_t)&lA[0][0];
  unsigned baseB = (unsigned)(uintptr_t)&lB[0][0];
  const int arow = tid >> 1, ach = (tid & 1)*16;
  const int bn = tid >> 2,  bseg = (tid & 3)*8;
  const unsigned short* hb = Hact + (size_t)b*T_*D_;
  const unsigned short* Abase = W1p + (size_t)e*3*FF_*D_;

  auto issueA = [&](int tap, int k0, int buf) {
    unsigned la = baseA + (unsigned)((buf*128 + arow)*LDSA_S + ach)*2u;
    const unsigned short* g = Abase + (size_t)tap*FF_*D_ + (size_t)(m0 + arow)*D_ + k0 + ach;
    async_ld_b128(la,      g);
    async_ld_b128(la + 16, g + 8);
  };
  auto issueB = [&](int tap, int k0, int buf) {
    int t = n0 + bn + tap - 1;
    unsigned lb = baseB + (unsigned)((buf*64 + bn)*LDSB_S + bseg)*2u;
    if (t >= 0 && t < T_) {
      async_ld_b128(lb, hb + (size_t)t*D_ + k0 + bseg);
    } else {
      u32x4 z = {0,0,0,0};
      *(u32x4*)&lB[buf][bn*LDSB_S + bseg] = z;
    }
  };

  v8f acc[2][2] = {};
  issueA(0, 0, 0); issueB(0, 0, 0);
  async_wait0();
  __syncthreads();
  const int NS = 3*(D_/32);                 // 96 steps
  for (int st = 0; st < NS; ++st) {
    int cur = st & 1, nxt = cur ^ 1;
    if (st + 1 < NS) {
      int t2 = (st+1) >> 5, k2 = ((st+1) & 31) * 32;
      issueA(t2, k2, nxt); issueB(t2, k2, nxt);
    }
    const unsigned short* cA = lA[cur];
    const unsigned short* cB = lB[cur];
    v16bf a0 = load_fragA(cA, wm, 0, lane), a1 = load_fragA(cA, wm, 1, lane);
    v16bf b0 = load_fragB(cB, wn, 0, lane), b1 = load_fragB(cB, wn, 1, lane);
    acc[0][0] = wmma_bf16(a0, b0, acc[0][0]);
    acc[0][1] = wmma_bf16(a0, b1, acc[0][1]);
    acc[1][0] = wmma_bf16(a1, b0, acc[1][0]);
    acc[1][1] = wmma_bf16(a1, b1, acc[1][1]);
    if (st + 1 < NS) async_wait0();
    __syncthreads();
  }
  int g = lane >> 4, lr = lane & 15;
  const float* bias = eb1 + (size_t)e*FF_;
  for (int mr = 0; mr < 2; ++mr)
    for (int nf = 0; nf < 2; ++nf) {
      int m = m0 + wm*32 + mr*16 + g*8;
      int t = n0 + wn*32 + nf*16 + lr;
      f32x4 c0 = *(const f32x4*)&bias[m];
      f32x4 c1 = *(const f32x4*)&bias[m+4];
      float bb[8] = { c0[0],c0[1],c0[2],c0[3], c1[0],c1[1],c1[2],c1[3] };
      v8f v = acc[mr][nf];
      unsigned short o[8];
#pragma unroll
      for (int r = 0; r < 8; ++r) o[r] = f2bs(fmaxf(v[r] + bb[r], 0.0f));
      u32x4 u;
      u[0] = (unsigned)o[0] | ((unsigned)o[1] << 16);
      u[1] = (unsigned)o[2] | ((unsigned)o[3] << 16);
      u[2] = (unsigned)o[4] | ((unsigned)o[5] << 16);
      u[3] = (unsigned)o[6] | ((unsigned)o[7] << 16);
      *(u32x4*)&Y1[(((size_t)(e*B_ + b)*T_ + t)*FF_ + m)] = u;
    }
}

// =======================================================================
// Expert conv2 (FF->D, KW=3) + gate-weighted combine over all 4 experts
// =======================================================================
__global__ __launch_bounds__(256) void k_conv2(
    const unsigned short* __restrict__ W2p,  // [E][KW][D][FF]
    const unsigned short* __restrict__ Y1,   // [E][B][T][FF]
    const float* __restrict__ eb2,           // [E][D]
    const float* __restrict__ gates,         // [NTOK][E]
    float* __restrict__ MOE)                 // [NTOK][D]
{
  __shared__ unsigned short lA[2][128*LDSA_S];
  __shared__ unsigned short lB[2][64*LDSB_S];
  int tid = threadIdx.x, lane = tid & 31, wave = tid >> 5;
  int wm = wave & 3, wn = wave >> 2;
  int nt = blockIdx.x & 7, mt = blockIdx.x >> 3;
  int b = blockIdx.y;
  int n0 = nt*64, m0 = mt*128;
  int g = lane >> 4, lr = lane & 15;
  unsigned baseA = (unsigned)(uintptr_t)&lA[0][0];
  unsigned baseB = (unsigned)(uintptr_t)&lB[0][0];
  const int arow = tid >> 1, ach = (tid & 1)*16;
  const int bn = tid >> 2,  bseg = (tid & 3)*8;
  v8f outv[2][2] = {};
  for (int e = 0; e < 4; ++e) {
    const unsigned short* yb = Y1 + (size_t)(e*B_ + b)*T_*FF_;
    const unsigned short* Abase = W2p + (size_t)e*3*D_*FF_;
    auto issueA = [&](int tap, int k0, int buf) {
      unsigned la = baseA + (unsigned)((buf*128 + arow)*LDSA_S + ach)*2u;
      const unsigned short* gp = Abase + (size_t)tap*D_*FF_ + (size_t)(m0 + arow)*FF_ + k0 + ach;
      async_ld_b128(la,      gp);
      async_ld_b128(la + 16, gp + 8);
    };
    auto issueB = [&](int tap, int k0, int buf) {
      int t = n0 + bn + tap - 1;
      unsigned lb = baseB + (unsigned)((buf*64 + bn)*LDSB_S + bseg)*2u;
      if (t >= 0 && t < T_) {
        async_ld_b128(lb, yb + (size_t)t*FF_ + k0 + bseg);
      } else {
        u32x4 z = {0,0,0,0};
        *(u32x4*)&lB[buf][bn*LDSB_S + bseg] = z;
      }
    };
    v8f acc[2][2] = {};
    issueA(0, 0, 0); issueB(0, 0, 0);
    async_wait0();
    __syncthreads();
    const int NS = 3*(FF_/32);              // 192 steps
    for (int st = 0; st < NS; ++st) {
      int cur = st & 1, nxt = cur ^ 1;
      if (st + 1 < NS) {
        int t2 = (st+1) >> 6, k2 = ((st+1) & 63) * 32;
        issueA(t2, k2, nxt); issueB(t2, k2, nxt);
      }
      const unsigned short* cA = lA[cur];
      const unsigned short* cB = lB[cur];
      v16bf a0 = load_fragA(cA, wm, 0, lane), a1 = load_fragA(cA, wm, 1, lane);
      v16bf b0 = load_fragB(cB, wn, 0, lane), b1 = load_fragB(cB, wn, 1, lane);
      acc[0][0] = wmma_bf16(a0, b0, acc[0][0]);
      acc[0][1] = wmma_bf16(a0, b1, acc[0][1]);
      acc[1][0] = wmma_bf16(a1, b0, acc[1][0]);
      acc[1][1] = wmma_bf16(a1, b1, acc[1][1]);
      if (st + 1 < NS) async_wait0();
      __syncthreads();
    }
    const float* bias = eb2 + (size_t)e*D_;
    for (int mr = 0; mr < 2; ++mr)
      for (int nf = 0; nf < 2; ++nf) {
        int m = m0 + wm*32 + mr*16 + g*8;
        int t = n0 + wn*32 + nf*16 + lr;
        float gv = gates[((size_t)b*T_ + t)*4 + e];
        f32x4 c0 = *(const f32x4*)&bias[m];
        f32x4 c1 = *(const f32x4*)&bias[m+4];
        float bb[8] = { c0[0],c0[1],c0[2],c0[3], c1[0],c1[1],c1[2],c1[3] };
#pragma unroll
        for (int r = 0; r < 8; ++r) outv[mr][nf][r] += gv * (acc[mr][nf][r] + bb[r]);
      }
  }
  for (int mr = 0; mr < 2; ++mr)
    for (int nf = 0; nf < 2; ++nf) {
      int m = m0 + wm*32 + mr*16 + g*8;
      int t = n0 + wn*32 + nf*16 + lr;
      float* o = &MOE[((size_t)b*T_ + t)*D_ + m];
      v8f v = outv[mr][nf];
      f32x4 o0 = { v[0],v[1],v[2],v[3] }, o1 = { v[4],v[5],v[6],v[7] };
      *(f32x4*)&o[0] = o0;
      *(f32x4*)&o[4] = o1;
    }
}

// =======================================================================
// LayerNorm (in-place on feat region) + feat_norm + bf16 feat for outproj
// =======================================================================
__global__ __launch_bounds__(256) void k_layernorm(
    float* __restrict__ feat, float* __restrict__ feat_norm,
    unsigned short* __restrict__ featb,
    const float* __restrict__ lg, const float* __restrict__ lb)
{
  int bt = blockIdx.x, tid = threadIdx.x;
  float* row = feat + (size_t)bt*D_;
  float x[4], s = 0.f, sq = 0.f;
  for (int i = 0; i < 4; ++i) { x[i] = row[tid + i*256]; s += x[i]; sq += x[i]*x[i]; }
  __shared__ float r1[256], r2[256];
  r1[tid] = s; r2[tid] = sq; __syncthreads();
  for (int st = 128; st > 0; st >>= 1) {
    if (tid < st) { r1[tid] += r1[tid+st]; r2[tid] += r2[tid+st]; } __syncthreads();
  }
  float mu = r1[0] / D_;
  float var = r2[0] / D_ - mu*mu;
  float rs = rsqrtf(var + 1e-6f);
  __syncthreads();
  float f[4], n2 = 0.f;
  for (int i = 0; i < 4; ++i) {
    int d = tid + i*256;
    f[i] = (x[i]-mu)*rs*lg[d] + lb[d];
    n2 += f[i]*f[i];
  }
  r1[tid] = n2; __syncthreads();
  for (int st = 128; st > 0; st >>= 1) { if (tid < st) r1[tid] += r1[tid+st]; __syncthreads(); }
  float inv = 1.0f / fmaxf(sqrtf(r1[0]), 1e-12f);
  for (int i = 0; i < 4; ++i) {
    int d = tid + i*256;
    row[d] = f[i];
    feat_norm[(size_t)bt*D_ + d] = f[i]*inv;
    featb[(size_t)bt*D_ + d] = f2bs(f[i]);
  }
}

// =======================================================================
// Out projection: logits[t][c] = feat[t]·out_w[c] + b[c]   (M=1200 guarded)
// =======================================================================
__global__ __launch_bounds__(256) void k_gemm_out(
    const unsigned short* __restrict__ Wb,  // [CLS][D] bf16
    const unsigned short* __restrict__ Fb,  // [NTOK][D] bf16
    const float* __restrict__ bias,         // [CLS]
    float* __restrict__ L)                  // [NTOK][CLS]
{
  __shared__ unsigned short lA[2][128*LDSA_S];
  __shared__ unsigned short lB[2][64*LDSB_S];
  int tid = threadIdx.x, lane = tid & 31, wave = tid >> 5;
  int wm = wave & 3, wn = wave >> 2;
  int n0 = blockIdx.x * 64;
  int m0 = blockIdx.y * 128;
  unsigned baseA = (unsigned)(uintptr_t)&lA[0][0];
  unsigned baseB = (unsigned)(uintptr_t)&lB[0][0];
  const int arow = tid >> 1, ach = (tid & 1)*16;
  const int bn = tid >> 2,  bseg = (tid & 3)*8;
  const int agr = m0 + arow;

  auto issueA = [&](int k0, int buf) {
    unsigned la = baseA + (unsigned)((buf*128 + arow)*LDSA_S + ach)*2u;
    if (agr < CLS_) {
      const unsigned short* gp = &Wb[(size_t)agr*D_ + k0 + ach];
      async_ld_b128(la,      gp);
      async_ld_b128(la + 16, gp + 8);
    } else {
      u32x4 z = {0,0,0,0};
      *(u32x4*)&lA[buf][arow*LDSA_S + ach]     = z;
      *(u32x4*)&lA[buf][arow*LDSA_S + ach + 8] = z;
    }
  };
  auto issueB = [&](int k0, int buf) {
    unsigned lb = baseB + (unsigned)((buf*64 + bn)*LDSB_S + bseg)*2u;
    async_ld_b128(lb, &Fb[(size_t)(n0 + bn)*D_ + k0 + bseg]);
  };

  v8f acc[2][2] = {};
  issueA(0, 0); issueB(0, 0);
  async_wait0();
  __syncthreads();
  const int NS = D_/32;
  for (int st = 0; st < NS; ++st) {
    int cur = st & 1, nxt = cur ^ 1;
    if (st + 1 < NS) { issueA((st+1)*32, nxt); issueB((st+1)*32, nxt); }
    const unsigned short* cA = lA[cur];
    const unsigned short* cB = lB[cur];
    v16bf a0 = load_fragA(cA, wm, 0, lane), a1 = load_fragA(cA, wm, 1, lane);
    v16bf b0 = load_fragB(cB, wn, 0, lane), b1 = load_fragB(cB, wn, 1, lane);
    acc[0][0] = wmma_bf16(a0, b0, acc[0][0]);
    acc[0][1] = wmma_bf16(a0, b1, acc[0][1]);
    acc[1][0] = wmma_bf16(a1, b0, acc[1][0]);
    acc[1][1] = wmma_bf16(a1, b1, acc[1][1]);
    if (st + 1 < NS) async_wait0();
    __syncthreads();
  }
  int g = lane >> 4, lr = lane & 15;
  for (int mr = 0; mr < 2; ++mr)
    for (int nf = 0; nf < 2; ++nf) {
      int m = m0 + wm*32 + mr*16 + g*8;
      if (m >= CLS_) continue;
      int n = n0 + wn*32 + nf*16 + lr;
      f32x4 c0 = *(const f32x4*)&bias[m];
      f32x4 c1 = *(const f32x4*)&bias[m+4];
      v8f v = acc[mr][nf];
      f32x4 o0 = { v[0]+c0[0], v[1]+c0[1], v[2]+c0[2], v[3]+c0[3] };
      f32x4 o1 = { v[4]+c1[0], v[5]+c1[1], v[6]+c1[2], v[7]+c1[3] };
      float* o = &L[(size_t)n*CLS_ + m];
      *(f32x4*)&o[0] = o0;
      *(f32x4*)&o[4] = o1;
    }
}

// =======================================================================
// log-softmax over CLS, p = exp(logp)
// =======================================================================
__global__ __launch_bounds__(256) void k_softmax(
    const float* __restrict__ logits, float* __restrict__ logp, float* __restrict__ p)
{
  int bt = blockIdx.x, tid = threadIdx.x;
  const float* row = logits + (size_t)bt*CLS_;
  __shared__ float r[256];
  float mx = -3.4e38f;
  for (int c = tid; c < CLS_; c += 256) mx = fmaxf(mx, row[c]);
  r[tid] = mx; __syncthreads();
  for (int s = 128; s > 0; s >>= 1) { if (tid < s) r[tid] = fmaxf(r[tid], r[tid+s]); __syncthreads(); }
  mx = r[0]; __syncthreads();
  float se = 0.f;
  for (int c = tid; c < CLS_; c += 256) se += __expf(row[c]-mx);
  r[tid] = se; __syncthreads();
  for (int s = 128; s > 0; s >>= 1) { if (tid < s) r[tid] += r[tid+s]; __syncthreads(); }
  float lse = mx + __logf(r[0]);
  for (int c = tid; c < CLS_; c += 256) {
    float lp = row[c] - lse;
    logp[(size_t)bt*CLS_ + c] = lp;
    p[(size_t)bt*CLS_ + c] = __expf(lp);
  }
}

__global__ void k_aux(const float* __restrict__ aux, float* __restrict__ out) {
  if (threadIdx.x == 0 && blockIdx.x == 0) {
    float s = 0.f;
    for (int e = 0; e < 4; ++e) {
      float imp  = aux[e]   / (float)NTOK;
      float frac = aux[4+e] / (float)NTOK;
      s += frac * imp;
    }
    out[0] = 4.0f * s;
  }
}

// =======================================================================
extern "C" void kernel_launch(void* const* d_in, const int* in_sizes, int n_in,
                              void* d_out, int out_size, void* d_ws, size_t ws_size,
                              hipStream_t stream) {
  const float* x      = (const float*)d_in[0];
  const int*   mask   = (const int*)  d_in[1];
  const float* fc1_w  = (const float*)d_in[2];
  const float* fc1_b  = (const float*)d_in[3];
  const float* bn_g   = (const float*)d_in[4];
  const float* bn_b   = (const float*)d_in[5];
  const float* pe     = (const float*)d_in[6];
  const float* gate_w = (const float*)d_in[7];
  const float* ew1    = (const float*)d_in[8];
  const float* eb1    = (const float*)d_in[9];
  const float* ew2    = (const float*)d_in[10];
  const float* eb2    = (const float*)d_in[11];
  const float* ln_g   = (const float*)d_in[12];
  const float* ln_b   = (const float*)d_in[13];
  const float* out_w  = (const float*)d_in[14];
  const float* out_b  = (const float*)d_in[15];
  (void)in_sizes; (void)n_in; (void)out_size; (void)ws_size;

  // ---- workspace carve-up ----
  char* ws = (char*)d_ws;
  size_t off = 0;
  auto carve = [&](size_t bytes) -> char* {
    char* p = ws + off; off += (bytes + 255) & ~(size_t)255; return p;
  };
  float*          h_buf  = (float*)         carve(sizeof(float)*(size_t)NTOK*D_);
  unsigned short* hact   = (unsigned short*)carve(2ull*NTOK*D_);
  unsigned short* fc1wb  = (unsigned short*)carve(2ull*D_*D_);
  unsigned short* outwb  = (unsigned short*)carve(2ull*CLS_*D_);
  unsigned short* w1p    = (unsigned short*)carve(2ull*E_*KW_*FF_*D_);
  unsigned short* w2p    = (unsigned short*)carve(2ull*E_*KW_*D_*FF_);
  unsigned short* y1     = (unsigned short*)carve(2ull*E_*B_*T_*FF_);
  unsigned short* featb  = (unsigned short*)carve(2ull*NTOK*D_);
  float*          gates  = (float*)         carve(sizeof(float)*(size_t)NTOK*E_);
  float*          stats  = (float*)         carve(sizeof(float)*2*D_);
  float*          cnt    = (float*)         carve(sizeof(float)*16);
  float*          auxacc = (float*)         carve(sizeof(float)*16);

  // ---- output regions ----
  float* feat      = (float*)d_out;                         // [NTOK][D] (moe_out first, LN in place)
  float* feat_norm = feat      + (size_t)NTOK*D_;
  float* logits    = feat_norm + (size_t)NTOK*D_;
  float* logp      = logits    + (size_t)NTOK*CLS_;
  float* pbuf      = logp      + (size_t)NTOK*CLS_;
  float* auxout    = pbuf      + (size_t)NTOK*CLS_;

  hipMemsetAsync(stats,  0, sizeof(float)*2*D_, stream);
  hipMemsetAsync(auxacc, 0, sizeof(float)*16,   stream);

  // ---- weight packing ----
  k_cvt_bf16<<<(D_*D_)/256, 256, 0, stream>>>(fc1_w, fc1wb, D_*D_);
  k_cvt_bf16<<<(CLS_*D_)/256, 256, 0, stream>>>(out_w, outwb, CLS_*D_);
  k_pack_w1<<<(E_*KW_*FF_*D_)/256, 256, 0, stream>>>(ew1, w1p);
  k_pack_w2<<<(E_*KW_*D_*FF_)/256, 256, 0, stream>>>(ew2, w2p);

  // ---- fc1 + masked BN + ReLU + PE ----
  k_gemm_fc1<<<dim3(NTOK/64, D_/128), 256, 0, stream>>>(fc1wb, x, fc1_b, h_buf);
  k_maskcnt<<<1, 256, 0, stream>>>(mask, cnt);
  k_stats<<<dim3(D_/256, NTOK/256), 256, 0, stream>>>(h_buf, mask, stats);
  k_normapply<<<(NTOK*D_)/256, 256, 0, stream>>>(h_buf, mask, stats, cnt, bn_g, bn_b, pe, hact);

  // ---- gating ----
  k_gate<<<NTOK, 256, 0, stream>>>(hact, gate_w, gates, auxacc);

  // ---- MoE experts (WMMA conv GEMMs) ----
  k_conv1<<<dim3((FF_/128)*(T_/64), B_, E_), 256, 0, stream>>>(w1p, hact, eb1, y1);
  k_conv2<<<dim3((D_/128)*(T_/64), B_), 256, 0, stream>>>(w2p, y1, eb2, gates, feat);

  // ---- LayerNorm + heads ----
  k_layernorm<<<NTOK, 256, 0, stream>>>(feat, feat_norm, featb, ln_g, ln_b);
  k_gemm_out<<<dim3(NTOK/64, (CLS_ + 127)/128), 256, 0, stream>>>(outwb, featb, out_b, logits);
  k_softmax<<<NTOK, 256, 0, stream>>>(logits, logp, pbuf);
  k_aux<<<1, 32, 0, stream>>>(auxacc, auxout);
}
